// DynamicEncoderLayer_5712306504275
// MI455X (gfx1250) — matrix operand order repaired and checked by reading, not compiled, not run on hardware
//
#include <hip/hip_runtime.h>
#include <hip/hip_bf16.h>

// ---------------------------------------------------------------------------
// CDNA5 (gfx1250, wave32) encoder layer.
// GEMMs: v_wmma_f32_16x16x32_bf16, 128x64 block tile, 32x32 wave tile,
// K-step 64 (8 WMMA / barrier pair / wave), double-buffered LDS fed by
// GLOBAL_LOAD_ASYNC_TO_LDS_B128 (ASYNCcnt) — the CDNA5 VGPR-bypass path.
// ---------------------------------------------------------------------------

typedef __attribute__((ext_vector_type(16))) __bf16 v16bf;
typedef __attribute__((ext_vector_type(8)))  float  v8f;
typedef unsigned short ushort_t;

union FragAB {
    v16bf        v;
    unsigned int u[8];
};

__device__ __forceinline__ ushort_t f2bf(float x) {
    unsigned int u = __float_as_uint(x);
    unsigned int r = u + 0x7FFFu + ((u >> 16) & 1u);   // round-to-nearest-even
    return (ushort_t)(r >> 16);
}

__device__ __forceinline__ unsigned ldsOffsetOf(const void* p) {
    // generic (flat) LDS pointer: addr[31:0] is the LDS byte offset
    return (unsigned)(unsigned long long)p;
}

// async 16B copy global -> LDS, tracked by ASYNCcnt (bypasses VGPRs)
__device__ __forceinline__ void asyncCopy16(unsigned ldsOff, const ushort_t* g) {
    asm volatile("global_load_async_to_lds_b128 %0, %1, off"
                 :: "v"(ldsOff), "v"(g) : "memory");
}
__device__ __forceinline__ void waitAsync6() {
    asm volatile("s_wait_asynccnt 0x6" ::: "memory");
}
__device__ __forceinline__ void waitAsync0() {
    asm volatile("s_wait_asynccnt 0x0" ::: "memory");
}

#define WMMA_BF16(AF, BF, ACC) \
    __builtin_amdgcn_wmma_f32_16x16x32_bf16(false, (AF).v, false, (BF).v, (short)0, (ACC), false, false)

// ---------------------------------------------------------------------------
// GEMM:  C[z] = alpha * A[z] x B[z]^T (+ bias) (relu?) , bf16 inputs.
//   A: [M x K] bf16 row-major (lda, batch stride sA)
//   B: [N x K] bf16 row-major (ldb, batch stride sB)   <- always "NK"
//   C: [M x N] f32 or bf16 (flags&4) row-major (ldc, batch stride sC)
// Block = 256 threads (8 waves, 4x2).  Tile 128(M) x 64(N), K-step 64.
// Each wave: 32x32 via 4 accumulators; 8 WMMA per K-tile.
// Requires M%128==0, N%64==0, K%64==0, lda/ldb%8==0.
// flags: 2 = relu, 4 = bf16 output.
// ---------------------------------------------------------------------------
#define PIT 72   // LDS row pitch (halfwords): 144B = 16B-aligned, bank-clean

__global__ __launch_bounds__(256) void gemm_wmma_bf16(
    const ushort_t* __restrict__ A, const ushort_t* __restrict__ B,
    void* __restrict__ Cv, const float* __restrict__ bias,
    int M, int N, int K, int lda, int ldb, int ldc,
    long long sA, long long sB, long long sC,
    float alpha, int flags)
{
    (void)M; (void)N;
    const int tid  = threadIdx.x;
    const int wave = tid >> 5;
    const int lane = tid & 31;
    const int half = lane >> 4;
    const int l    = lane & 15;
    const int wmO  = (wave >> 1) * 32;    // wave tile row offset (0/32/64/96)
    const int wnO  = (wave & 1)  * 32;    // wave tile col offset (0/32)
    const int m0   = blockIdx.y * 128;
    const int n0   = blockIdx.x * 64;
    const long long z = blockIdx.z;

    A += z * sA;  B += z * sB;

    __shared__ ushort_t As[2][128 * PIT];   // double-buffered 128 x 64 bf16
    __shared__ ushort_t Bs[2][64 * PIT];    // double-buffered  64 x 64 bf16

    // per-thread staging: 16B chunks, contiguous-K
    const int rA = tid >> 3;               // 0..31 (A rows rA, rA+32, +64, +96)
    const int kc = (tid & 7) * 8;          // K chunk within 64
    const unsigned asB = ldsOffsetOf(&As[0][0]);
    const unsigned bsB = ldsOffsetOf(&Bs[0][0]);

    FragAB fa0, fa1, fb0, fb1;
    v8f a00 = {}, a01 = {}, a10 = {}, a11 = {};

    const int nt = K >> 6;                 // K / 64 tiles

    // issue one K-tile (6 async 16B ops per thread: 4 for A, 2 for B)
    auto issueTile = [&](int k0, int buf) {
#pragma unroll
        for (int it = 0; it < 4; ++it) {
            const int m = rA + it * 32;
            asyncCopy16(asB + (unsigned)(buf * 128 * PIT + m * PIT + kc) * 2,
                        A + (size_t)(m0 + m) * lda + k0 + kc);
        }
#pragma unroll
        for (int it = 0; it < 2; ++it) {
            const int n = rA + it * 32;
            asyncCopy16(bsB + (unsigned)(buf * 64 * PIT + n * PIT + kc) * 2,
                        B + (size_t)(n0 + n) * ldb + k0 + kc);
        }
    };

    issueTile(0, 0);                       // prologue: tile 0 -> buffer 0

    for (int t = 0; t < nt; ++t) {
        if (t + 1 < nt) {
            issueTile((t + 1) << 6, (t + 1) & 1);
            if (t + 2 < nt) {              // prefetch 2 tiles ahead
                __builtin_prefetch(A + (size_t)(m0 + rA) * lda + ((t + 2) << 6) + kc, 0, 1);
                __builtin_prefetch(B + (size_t)(n0 + rA) * ldb + ((t + 2) << 6) + kc, 0, 1);
            }
            waitAsync6();                  // tile t landed (in-order), t+1 in flight
        } else {
            waitAsync0();
        }
        __syncthreads();

        const int buf = t & 1;
        const ushort_t* ar0 = &As[buf][(wmO + l)      * PIT];
        const ushort_t* ar1 = &As[buf][(wmO + 16 + l) * PIT];
        const ushort_t* br0 = &Bs[buf][(wnO + l)      * PIT];
        const ushort_t* br1 = &Bs[buf][(wnO + 16 + l) * PIT];

#pragma unroll
        for (int ks = 0; ks < 2; ++ks) {   // two 32-wide K slices
            const int o = ks * 32;
#pragma unroll
            for (int j = 0; j < 4; ++j) {
                fa0.u[j]     = *(const unsigned int*)(ar0 + o + half * 16 + 2 * j);
                fa0.u[4 + j] = *(const unsigned int*)(ar0 + o + half * 16 + 8 + 2 * j);
                fa1.u[j]     = *(const unsigned int*)(ar1 + o + half * 16 + 2 * j);
                fa1.u[4 + j] = *(const unsigned int*)(ar1 + o + half * 16 + 8 + 2 * j);
            }
#pragma unroll
            for (int j = 0; j < 8; ++j) {
                fb0.u[j] = *(const unsigned int*)(br0 + o + half * 16 + 2 * j);
                fb1.u[j] = *(const unsigned int*)(br1 + o + half * 16 + 2 * j);
            }
            a00 = WMMA_BF16(fa0, fb0, a00);
            a01 = WMMA_BF16(fa0, fb1, a01);
            a10 = WMMA_BF16(fa1, fb0, a10);
            a11 = WMMA_BF16(fa1, fb1, a11);
        }
        __syncthreads();
    }

    // ---- epilogue ----
    float*    Cf = (float*)Cv    + z * sC;
    ushort_t* Cb = (ushort_t*)Cv + z * sC;
    const v8f accs[2][2] = { { a00, a01 }, { a10, a11 } };
#pragma unroll
    for (int ti = 0; ti < 2; ++ti) {
#pragma unroll
        for (int tj = 0; tj < 2; ++tj) {
            const int n  = n0 + wnO + tj * 16 + l;
            const float bv = bias ? bias[n] : 0.0f;
            v8f acc = accs[ti][tj];
#pragma unroll
            for (int i = 0; i < 8; ++i) {
                const int m = m0 + wmO + ti * 16 + half * 8 + i;
                float val = acc[i] * alpha + bv;
                if (flags & 2) val = fmaxf(val, 0.0f);
                if (flags & 4) Cb[(size_t)m * ldc + n] = f2bf(val);
                else           Cf[(size_t)m * ldc + n] = val;
            }
        }
    }
}

// ---------------------------------------------------------------------------
// Flat f32 -> bf16 convert (4 elements/thread).  n % 1024 == 0.
// ---------------------------------------------------------------------------
__global__ __launch_bounds__(256) void cvt_bf16(
    const float* __restrict__ src, ushort_t* __restrict__ dst)
{
    const size_t i = ((size_t)blockIdx.x * 256 + threadIdx.x) * 4;
    const float4 f = *reinterpret_cast<const float4*>(src + i);
    ushort4 o;
    o.x = f2bf(f.x); o.y = f2bf(f.y); o.z = f2bf(f.z); o.w = f2bf(f.w);
    *reinterpret_cast<ushort4*>(dst + i) = o;
}

// ---------------------------------------------------------------------------
// Batched transpose + convert:  dst[z][c][r] (bf16) = src[z][r][c] (f32).
// grid (C/32, R/32, batch), block (32,8).  R%32==0, C%32==0.
// ---------------------------------------------------------------------------
__global__ __launch_bounds__(256) void transpose_cvt(
    const float* __restrict__ src, ushort_t* __restrict__ dst,
    int R, int C, long long sSrc, long long sDst)
{
    __shared__ ushort_t t[32][33];
    const int c0 = blockIdx.x * 32, r0 = blockIdx.y * 32;
    const int tx = threadIdx.x, ty = threadIdx.y;
    const long long z = blockIdx.z;
    src += z * sSrc;  dst += z * sDst;
#pragma unroll
    for (int j = 0; j < 4; ++j)
        t[ty + j * 8][tx] = f2bf(src[(size_t)(r0 + ty + j * 8) * C + c0 + tx]);
    __syncthreads();
#pragma unroll
    for (int j = 0; j < 4; ++j)
        dst[(size_t)(c0 + ty + j * 8) * R + r0 + tx] = t[tx][ty + j * 8];
}

// ---------------------------------------------------------------------------
// In-place f32 softmax over rows of 1024 + bf16 copy for the next GEMM.
// ---------------------------------------------------------------------------
__global__ __launch_bounds__(256) void softmax1024(
    float* __restrict__ attn, ushort_t* __restrict__ attn_bf)
{
    __shared__ float red[256];
    const size_t base = (size_t)blockIdx.x * 1024;
    float* p = attn + base;
    ushort_t* pb = attn_bf + base;
    const int t = threadIdx.x;

    float x0 = p[t], x1 = p[t + 256], x2 = p[t + 512], x3 = p[t + 768];
    float m = fmaxf(fmaxf(x0, x1), fmaxf(x2, x3));
    red[t] = m; __syncthreads();
    for (int s = 128; s; s >>= 1) {
        if (t < s) red[t] = fmaxf(red[t], red[t + s]);
        __syncthreads();
    }
    m = red[0]; __syncthreads();

    x0 = __expf(x0 - m); x1 = __expf(x1 - m);
    x2 = __expf(x2 - m); x3 = __expf(x3 - m);
    red[t] = x0 + x1 + x2 + x3; __syncthreads();
    for (int s = 128; s; s >>= 1) {
        if (t < s) red[t] += red[t + s];
        __syncthreads();
    }
    const float inv = 1.0f / red[0];
    x0 *= inv; x1 *= inv; x2 *= inv; x3 *= inv;
    p[t] = x0;        p[t + 256] = x1;        p[t + 512] = x2;        p[t + 768] = x3;
    pb[t] = f2bf(x0); pb[t + 256] = f2bf(x1); pb[t + 512] = f2bf(x2); pb[t + 768] = f2bf(x3);
}

// ---------------------------------------------------------------------------
// out = LN(a + b)*g + beta over rows of 1024; optional bf16 copy.
// ---------------------------------------------------------------------------
__global__ __launch_bounds__(256) void ln_add1024(
    const float* __restrict__ a, const float* __restrict__ b,
    const float* __restrict__ g, const float* __restrict__ be,
    float* __restrict__ o, ushort_t* __restrict__ ob)
{
    __shared__ float red[256];
    const size_t base = (size_t)blockIdx.x * 1024;
    const int t = threadIdx.x;

    float v0 = a[base + t]       + b[base + t];
    float v1 = a[base + t + 256] + b[base + t + 256];
    float v2 = a[base + t + 512] + b[base + t + 512];
    float v3 = a[base + t + 768] + b[base + t + 768];

    red[t] = v0 + v1 + v2 + v3; __syncthreads();
    for (int s = 128; s; s >>= 1) {
        if (t < s) red[t] += red[t + s];
        __syncthreads();
    }
    const float mean = red[0] * (1.0f / 1024.0f); __syncthreads();

    v0 -= mean; v1 -= mean; v2 -= mean; v3 -= mean;
    red[t] = v0 * v0 + v1 * v1 + v2 * v2 + v3 * v3; __syncthreads();
    for (int s = 128; s; s >>= 1) {
        if (t < s) red[t] += red[t + s];
        __syncthreads();
    }
    const float rstd = rsqrtf(red[0] * (1.0f / 1024.0f) + 1e-5f);

    const float r0 = v0 * rstd * g[t]       + be[t];
    const float r1 = v1 * rstd * g[t + 256] + be[t + 256];
    const float r2 = v2 * rstd * g[t + 512] + be[t + 512];
    const float r3 = v3 * rstd * g[t + 768] + be[t + 768];
    o[base + t] = r0; o[base + t + 256] = r1;
    o[base + t + 512] = r2; o[base + t + 768] = r3;
    if (ob) {
        ob[base + t] = f2bf(r0); ob[base + t + 256] = f2bf(r1);
        ob[base + t + 512] = f2bf(r2); ob[base + t + 768] = f2bf(r3);
    }
}

// ---------------------------------------------------------------------------
// Host orchestration.  B=4, S=1024, D=1024, H=16, d=64, F=4096, BH=64.
// Raw-reshape quirk: head (b,h) = contiguous 65536-float block as 1024x64.
// ---------------------------------------------------------------------------
extern "C" void kernel_launch(void* const* d_in, const int* in_sizes, int n_in,
                              void* d_out, int out_size, void* d_ws, size_t ws_size,
                              hipStream_t stream)
{
    (void)in_sizes; (void)n_in; (void)out_size; (void)ws_size;

    const float* k   = (const float*)d_in[0];
    const float* v   = (const float*)d_in[1];
    const float* q   = (const float*)d_in[2];
    const float* r   = (const float*)d_in[3];
    const float* Wf  = (const float*)d_in[4];
    const float* bf  = (const float*)d_in[5];
    const float* g1  = (const float*)d_in[6];
    const float* b1  = (const float*)d_in[7];
    const float* W1  = (const float*)d_in[8];
    const float* bw1 = (const float*)d_in[9];
    const float* W2  = (const float*)d_in[10];
    const float* bw2 = (const float*)d_in[11];
    const float* g2  = (const float*)d_in[12];
    const float* b2  = (const float*)d_in[13];

    float* out  = (float*)d_out;                       // [4,1024,1024]
    float* attn = out + (size_t)4 * 1024 * 1024;       // [64,1024,1024] f32

    // workspace partition (bytes, 256B aligned regions)
    char* w = (char*)d_ws;
    const size_t MB = 1024 * 1024;
    ushort_t* vb    = (ushort_t*)(w + 0);              // 8 MB  [64][1024x64]
    ushort_t* kb    = (ushort_t*)(w + 8  * MB);        // 8 MB
    ushort_t* qT    = (ushort_t*)(w + 16 * MB);        // 8 MB  [64][64x1024]
    ushort_t* WfT   = (ushort_t*)(w + 24 * MB);        // 2 MB  [1024x1024]
    ushort_t* W1T   = (ushort_t*)(w + 26 * MB);        // 8 MB  [4096x1024]
    ushort_t* W2T   = (ushort_t*)(w + 34 * MB);        // 8 MB  [1024x4096]
    ushort_t* ctxb  = (ushort_t*)(w + 42 * MB);        // 8 MB  [4096x1024]
    ushort_t* xb    = (ushort_t*)(w + 50 * MB);        // 8 MB
    ushort_t* h     = (ushort_t*)(w + 58 * MB);        // 32 MB [4096x4096]
    float*    tmp   = (float*)   (w + 90 * MB);        // 16 MB
    float*    x     = (float*)   (w + 106* MB);        // 16 MB
    float*    y     = tmp;                             // reuse (tmp dead)
    ushort_t* attnb = (ushort_t*)(w + 122* MB);        // 128 MB [64][1024x1024]

    const dim3 blk(256), blkT(32, 8);
    const long long HS = 65536;      // head block stride (elements)
    const long long AS = 1048576;    // attn per-head stride (elements)

    // ---- prepasses: convert / transpose-convert GEMM operands to bf16 ----
    cvt_bf16<<<4096, blk, 0, stream>>>(v, vb);
    cvt_bf16<<<4096, blk, 0, stream>>>(k, kb);
    transpose_cvt<<<dim3(2, 32, 64),  blkT, 0, stream>>>(q,  qT,  1024, 64,   HS, HS);
    transpose_cvt<<<dim3(32, 32, 1),  blkT, 0, stream>>>(Wf, WfT, 1024, 1024, 0, 0);
    transpose_cvt<<<dim3(128, 32, 1), blkT, 0, stream>>>(W1, W1T, 1024, 4096, 0, 0);
    transpose_cvt<<<dim3(32, 128, 1), blkT, 0, stream>>>(W2, W2T, 4096, 1024, 0, 0);

    // 1) scores[z] = (v_head . k_head^T) * sqrt(64)  -> f32 attn (d_out)
    gemm_wmma_bf16<<<dim3(16, 8, 64), blk, 0, stream>>>(
        vb, kb, attn, nullptr, 1024, 1024, 64, 64, 64, 1024,
        HS, HS, AS, 8.0f, 0);

    // 2) softmax rows; writes f32 attn (output) + bf16 copy for ctx GEMM
    softmax1024<<<65536, blk, 0, stream>>>(attn, attnb);

    // 3) ctx[z] = attn[z] . q_head[z]  -> bf16 ctx
    gemm_wmma_bf16<<<dim3(1, 8, 64), blk, 0, stream>>>(
        attnb, qT, ctxb, nullptr, 1024, 64, 1024, 1024, 1024, 64,
        AS, HS, HS, 1.0f, /*bf16 out*/4);

    // 4) tmp = ctx . Wf + bf   (ctx raw-flattens to [4096x1024])
    gemm_wmma_bf16<<<dim3(16, 32, 1), blk, 0, stream>>>(
        ctxb, WfT, tmp, bf, 4096, 1024, 1024, 1024, 1024, 1024,
        0, 0, 0, 1.0f, 0);

    // 5) x = LN(r + tmp)*g1 + b1  (+ bf16 copy)
    ln_add1024<<<4096, blk, 0, stream>>>(r, tmp, g1, b1, x, xb);

    // 6) h = relu(x . W1 + bw1) -> bf16
    gemm_wmma_bf16<<<dim3(64, 32, 1), blk, 0, stream>>>(
        xb, W1T, h, bw1, 4096, 4096, 1024, 1024, 1024, 4096,
        0, 0, 0, 1.0f, /*relu|bf16*/6);

    // 7) y = h . W2 + bw2
    gemm_wmma_bf16<<<dim3(16, 32, 1), blk, 0, stream>>>(
        h, W2T, y, bw2, 4096, 1024, 4096, 4096, 4096, 1024,
        0, 0, 0, 1.0f, 0);

    // 8) output = LN(x + y)*g2 + b2
    ln_add1024<<<4096, blk, 0, stream>>>(x, y, g2, b2, out, nullptr);
}
